// Mamba2Block_47115791237288
// MI455X (gfx1250) — matrix operand order repaired and checked by reading, not compiled, not run on hardware
//
#include <hip/hip_runtime.h>

typedef __attribute__((ext_vector_type(16))) __bf16 bf16x16;
typedef __attribute__((ext_vector_type(8)))  __bf16 bf16x8;
typedef __attribute__((ext_vector_type(8)))  float  f32x8;

#define D_MODEL   512
#define D_STATE   64
#define D_INNER   1024
#define NHEADS    16
#define HEADDIM   64
#define CONV_DIM  1152
#define D_IN_PROJ 2192
#define ZXP       2304    /* zx row pitch, padded to multiple of 128 */
#define BATCH     4
#define SEQLEN    4096
#define CHUNK     64
#define NCHUNK    64
#define NTOK      16384

// ---------------- WMMA fragment helpers (CDNA5 16-bit layouts, wave32) -------

// A-matrix 16x32 (MxK), 16-bit: lanes 0-15 row M=lane hold K {0..7, 16..23},
// lanes 16-31 row M=lane-16 hold K {8..15, 24..31}; 2 halves per VGPR.
__device__ __forceinline__ bf16x16 frag_a(const __bf16* base, int ld, int row0, int k0) {
  int lane = (int)(threadIdx.x & 31);
  const __bf16* p = base + (size_t)(row0 + (lane & 15)) * ld + k0 + ((lane < 16) ? 0 : 8);
  bf16x8 lo = *reinterpret_cast<const bf16x8*>(p);
  bf16x8 hi = *reinterpret_cast<const bf16x8*>(p + 16);
  return __builtin_shufflevector(lo, hi, 0,1,2,3,4,5,6,7,8,9,10,11,12,13,14,15);
}

// B-matrix 32x16 (KxN), 16-bit, from N-major ("transposed") storage baseT[n][k]:
// lanes 0-15 col N=lane hold K {0..7, 16..23}, lanes 16-31 col N=lane-16 hold K {8..15, 24..31}.
__device__ __forceinline__ bf16x16 frag_b(const __bf16* baseT, int ld, int col0, int k0) {
  int lane = (int)(threadIdx.x & 31);
  const __bf16* p = baseT + (size_t)(col0 + (lane & 15)) * ld + k0 + ((lane < 16) ? 0 : 8);
  bf16x8 lo = *reinterpret_cast<const bf16x8*>(p);
  bf16x8 hi = *reinterpret_cast<const bf16x8*>(p + 16);
  return __builtin_shufflevector(lo, hi, 0,1,2,3,4,5,6,7,8,9,10,11,12,13,14,15);
}

__device__ __forceinline__ f32x8 wmma_bf16(bf16x16 a, bf16x16 b, f32x8 c) {
  return __builtin_amdgcn_wmma_f32_16x16x32_bf16(false, a, false, b, (short)0, c, false, false);
}

// ---- gfx1250 async memory->LDS copy (ASYNCcnt-tracked, no VGPR round-trip) ---
// dst: 32-bit LDS byte address (low 32 bits of generic LDS pointer);
// src: 64-bit global address in a VGPR pair.
__device__ __forceinline__ void async_copy_b128(void* lds_dst, const void* gsrc) {
  unsigned laddr = (unsigned)(size_t)lds_dst;
  unsigned long long gaddr = (unsigned long long)(size_t)gsrc;
  asm volatile("global_load_async_to_lds_b128 %0, %1, off"
               :: "v"(laddr), "v"(gaddr) : "memory");
}
__device__ __forceinline__ void wait_async_all() {
  asm volatile("s_wait_asynccnt 0x0" ::: "memory");
}

// C/D 16x16 f32 fragment coords: col = lane&15 ; row = vr + (lane<16 ? 0 : 8).

// ---------------- weight prep: cast f32 -> bf16, transpose to N-major, pad ----

__global__ void castT_kernel(const float* __restrict__ W, __bf16* __restrict__ WT,
                             int K, int N, long total) {
  long idx = (long)blockIdx.x * 256 + threadIdx.x;
  if (idx >= total) return;
  int k = (int)(idx % K);
  long n = idx / K;
  WT[idx] = (n < N) ? (__bf16)W[(size_t)k * N + n] : (__bf16)0.0f;
}

// ---------------- elementwise kernels ----------------------------------------

__global__ void ln_kernel(const float* __restrict__ x, const float* __restrict__ w,
                          const float* __restrict__ b, __bf16* __restrict__ hbf) {
  __shared__ float red[256];
  int row = blockIdx.x, tid = threadIdx.x;
  const float* xr = x + (size_t)row * D_MODEL;
  float v0 = xr[tid], v1 = xr[tid + 256];
  red[tid] = v0 + v1; __syncthreads();
  for (int off = 128; off > 0; off >>= 1) { if (tid < off) red[tid] += red[tid + off]; __syncthreads(); }
  float mu = red[0] * (1.0f / D_MODEL);
  __syncthreads();
  red[tid] = (v0 - mu) * (v0 - mu) + (v1 - mu) * (v1 - mu); __syncthreads();
  for (int off = 128; off > 0; off >>= 1) { if (tid < off) red[tid] += red[tid + off]; __syncthreads(); }
  float inv = rsqrtf(red[0] * (1.0f / D_MODEL) + 1e-5f);
  hbf[(size_t)row * D_MODEL + tid]       = (__bf16)((v0 - mu) * inv * w[tid]       + b[tid]);
  hbf[(size_t)row * D_MODEL + tid + 256] = (__bf16)((v1 - mu) * inv * w[tid + 256] + b[tid + 256]);
}

__global__ void conv_silu_kernel(const float* __restrict__ zx, const float* __restrict__ cw,
                                 const float* __restrict__ cb, float* __restrict__ xBCc) {
  size_t idx = (size_t)blockIdx.x * 256 + threadIdx.x;
  if (idx >= (size_t)NTOK * CONV_DIM) return;
  int ch = (int)(idx % CONV_DIM);
  size_t tok = idx / CONV_DIM;
  int l = (int)(tok % SEQLEN);
  size_t brow = tok - l;
  float acc = cb[ch];
  #pragma unroll
  for (int k = 0; k < 4; ++k) {
    int tl = l - 3 + k;
    if (tl >= 0) acc += zx[(brow + tl) * ZXP + D_INNER + ch] * cw[ch * 4 + k];
  }
  xBCc[idx] = acc / (1.0f + __expf(-acc));
}

__global__ void dt_acum_kernel(const float* __restrict__ zx, const float* __restrict__ dt_bias,
                               const float* __restrict__ A_log, float* __restrict__ dtArr,
                               float* __restrict__ Acum, float* __restrict__ cdec) {
  __shared__ float sv[CHUNK];
  int c = blockIdx.x, h = blockIdx.y, b = blockIdx.z, l = threadIdx.x;
  size_t tok = (size_t)b * SEQLEN + c * CHUNK + l;
  float raw = zx[tok * ZXP + (D_IN_PROJ - NHEADS) + h] + dt_bias[h];
  float dt  = (raw > 20.0f) ? raw : log1pf(__expf(raw));
  dtArr[tok * NHEADS + h] = dt;
  sv[l] = dt * (-__expf(A_log[h]));
  __syncthreads();
  float ac = 0.0f;
  for (int i = 0; i <= l; ++i) ac += sv[i];
  Acum[((size_t)(b * NHEADS + h) * NCHUNK + c) * CHUNK + l] = ac;
  if (l == CHUNK - 1) cdec[(size_t)(b * NHEADS + h) * NCHUNK + c] = __expf(ac);
}

__global__ void scan_kernel(const float* __restrict__ states, const float* __restrict__ cdec,
                            float* __restrict__ prevS) {
  int h = blockIdx.x, b = blockIdx.y, tid = threadIdx.x;
  float s[16];
  #pragma unroll
  for (int j = 0; j < 16; ++j) s[j] = 0.0f;
  for (int c = 0; c < NCHUNK; ++c) {
    size_t off = ((size_t)(b * NCHUNK + c) * NHEADS + h) * (HEADDIM * D_STATE);
    float cd = cdec[(size_t)(b * NHEADS + h) * NCHUNK + c];
    #pragma unroll
    for (int j = 0; j < 16; ++j) {
      int e = j * 256 + tid;
      prevS[off + e] = s[j];
      s[j] = s[j] * cd + states[off + e];
    }
  }
}

__global__ void gate_kernel(const float* __restrict__ y, const float* __restrict__ zx,
                            const float* __restrict__ rms_w, __bf16* __restrict__ gbf) {
  __shared__ float red[256];
  int row = blockIdx.x, tid = threadIdx.x;
  const float* yr = y  + (size_t)row * D_INNER;
  const float* zr = zx + (size_t)row * ZXP;
  float g[4]; float ss = 0.0f;
  #pragma unroll
  for (int j = 0; j < 4; ++j) {
    int i = tid + j * 256;
    float zv = zr[i];
    float gv = yr[i] * (zv / (1.0f + __expf(-zv)));
    g[j] = gv; ss += gv * gv;
  }
  red[tid] = ss; __syncthreads();
  for (int off = 128; off > 0; off >>= 1) { if (tid < off) red[tid] += red[tid + off]; __syncthreads(); }
  float scale = rsqrtf(red[0] * (1.0f / D_INNER) + 1e-5f);
  #pragma unroll
  for (int j = 0; j < 4; ++j) {
    int i = tid + j * 256;
    gbf[(size_t)row * D_INNER + i] = (__bf16)(g[j] * scale * rms_w[i]);
  }
}

// ---------------- WMMA GEMM: C = A[M,K] @ W[K,N] (+R) ------------------------
// WT is pre-transposed N-major (row n holds K contiguous halves), N padded to
// a multiple of 128 with zeros. K must be a multiple of 64. No bounds checks.

#define GTN  128
#define WLDP 72   /* LDS row pitch (halves): 64 data + 8 pad, 16B-aligned rows */
__global__ void gemm_bf16_wmma(const __bf16* __restrict__ A, const __bf16* __restrict__ WT,
                               const float* __restrict__ R, float* __restrict__ C,
                               int K, int ldc) {
  __shared__ __bf16 ldsW[GTN * WLDP];
  int m0 = blockIdx.y * 64, n0 = blockIdx.x * GTN;
  int tid = threadIdx.x, lane = tid & 31, w = tid >> 5;
  int wm = w & 3, wn = w >> 2;        // wave tile: rows m0+wm*16, cols n0+wn*64
  f32x8 acc[4] = {};
  for (int k0 = 0; k0 < K; k0 += 64) {
    // stage WT[n0..n0+128)[k0..k0+64) -> LDS via gfx1250 async DMA (ASYNCcnt)
    #pragma unroll
    for (int i = 0; i < 4; ++i) {
      int c2 = tid + i * 256;
      int n = c2 >> 3, seg = c2 & 7;
      async_copy_b128(&ldsW[n * WLDP + seg * 8],
                      &WT[(size_t)(n0 + n) * K + k0 + seg * 8]);
    }
    if (k0 + 64 < K)  // hint next A slice into cache (global_prefetch_b8)
      __builtin_prefetch((const void*)(A + (size_t)(m0 + wm * 16 + (lane & 15)) * K + k0 + 64), 0, 0);
    wait_async_all();
    __syncthreads();
    #pragma unroll
    for (int kk = 0; kk < 64; kk += 32) {
      bf16x16 af = frag_a(A, K, m0 + wm * 16, k0 + kk);
      #pragma unroll
      for (int t = 0; t < 4; ++t)
        acc[t] = wmma_bf16(af, frag_b(ldsW, WLDP, wn * 64 + t * 16, kk), acc[t]);
    }
    __syncthreads();
  }
  #pragma unroll
  for (int t = 0; t < 4; ++t) {
    int col = n0 + wn * 64 + t * 16 + (lane & 15);
    #pragma unroll
    for (int vr = 0; vr < 8; ++vr) {
      int row = m0 + wm * 16 + vr + ((lane < 16) ? 0 : 8);
      float v = acc[t][vr];
      if (R) v += R[(size_t)row * ldc + col];
      C[(size_t)row * ldc + col] = v;
    }
  }
}

// ---------------- SSD per-chunk kernel: Y_diag + states -----------------------

#define SLD 72
__global__ void ssd_chunk_kernel(const float* __restrict__ xBCc, const float* __restrict__ dtArr,
                                 const float* __restrict__ Acum, const float* __restrict__ Dh,
                                 float* __restrict__ y, float* __restrict__ states) {
  __shared__ __bf16 sC[64 * SLD], sB[64 * SLD], sXT[64 * SLD], sBTd[64 * SLD], sG[64 * SLD];
  __shared__ float sAc[64], sDec[64];
  int h = blockIdx.x, c = blockIdx.y, b = blockIdx.z, tid = threadIdx.x;
  const float* AcP = Acum + ((size_t)(b * NHEADS + h) * NCHUNK + c) * CHUNK;
  if (tid < 64) sAc[tid] = AcP[tid];
  __syncthreads();
  if (tid < 64) sDec[tid] = __expf(sAc[63] - sAc[tid]);
  __syncthreads();
  for (int i = tid; i < 4096; i += 256) {
    int l = i >> 6, j = i & 63;
    size_t tok = (size_t)b * SEQLEN + c * CHUNK + l;
    const float* xr = xBCc + tok * CONV_DIM;
    float cv = xr[D_INNER + D_STATE + j];
    float bv = xr[D_INNER + j];
    float xv = xr[h * HEADDIM + j];
    float dt = dtArr[tok * NHEADS + h];
    sC[l * SLD + j]   = (__bf16)cv;
    sB[l * SLD + j]   = (__bf16)bv;                 // s-major: B-operand of Cc@Bc^T
    sBTd[j * SLD + l] = (__bf16)(bv * sDec[l]);     // n-major: B-operand of states
    sXT[j * SLD + l]  = (__bf16)(xv * dt);          // p-major: Xd^T
  }
  __syncthreads();
  int lane = tid & 31, w = tid >> 5, mt = w & 3, np = w >> 2;
  // matmul1: G = Cc (l x n) @ Bc^T (n x s)
  f32x8 g[2] = {};
  for (int kk = 0; kk < 64; kk += 32) {
    bf16x16 af = frag_a(sC, SLD, mt * 16, kk);
    #pragma unroll
    for (int t = 0; t < 2; ++t)
      g[t] = wmma_bf16(af, frag_b(sB, SLD, (np * 2 + t) * 16, kk), g[t]);
  }
  #pragma unroll
  for (int t = 0; t < 2; ++t) {
    int s = (np * 2 + t) * 16 + (lane & 15);
    #pragma unroll
    for (int vr = 0; vr < 8; ++vr) {
      int l = mt * 16 + vr + ((lane < 16) ? 0 : 8);
      float v = (s <= l) ? g[t][vr] * __expf(sAc[l] - sAc[s]) : 0.0f;
      sG[l * SLD + s] = (__bf16)v;
    }
  }
  __syncthreads();
  // matmul2: Y_diag = G (l x s) @ Xd (s x p)   [B-operand p-major = sXT]
  f32x8 yd[2] = {};
  for (int kk = 0; kk < 64; kk += 32) {
    bf16x16 af = frag_a(sG, SLD, mt * 16, kk);
    #pragma unroll
    for (int t = 0; t < 2; ++t)
      yd[t] = wmma_bf16(af, frag_b(sXT, SLD, (np * 2 + t) * 16, kk), yd[t]);
  }
  float Dv = Dh[h];
  #pragma unroll
  for (int t = 0; t < 2; ++t) {
    int p = (np * 2 + t) * 16 + (lane & 15);
    #pragma unroll
    for (int vr = 0; vr < 8; ++vr) {
      int l = mt * 16 + vr + ((lane < 16) ? 0 : 8);
      size_t tok = (size_t)b * SEQLEN + c * CHUNK + l;
      float xv = xBCc[tok * CONV_DIM + h * HEADDIM + p];
      y[tok * D_INNER + h * HEADDIM + p] = yd[t][vr] + Dv * xv;
    }
  }
  // matmul3: states = Xd^T (p x l) @ (Bc*decay) (l x n)  [B-operand n-major = sBTd]
  f32x8 st[2] = {};
  for (int kk = 0; kk < 64; kk += 32) {
    bf16x16 af = frag_a(sXT, SLD, mt * 16, kk);
    #pragma unroll
    for (int t = 0; t < 2; ++t)
      st[t] = wmma_bf16(af, frag_b(sBTd, SLD, (np * 2 + t) * 16, kk), st[t]);
  }
  float* stp = states + ((size_t)(b * NCHUNK + c) * NHEADS + h) * (HEADDIM * D_STATE);
  #pragma unroll
  for (int t = 0; t < 2; ++t) {
    int n = (np * 2 + t) * 16 + (lane & 15);
    #pragma unroll
    for (int vr = 0; vr < 8; ++vr) {
      int p = mt * 16 + vr + ((lane < 16) ? 0 : 8);
      stp[p * D_STATE + n] = st[t][vr];
    }
  }
}

// ---------------- SSD Y_off kernel: y += (Cc*e^Acum) @ prev^T ------------------

__global__ void ssd_yoff_kernel(const float* __restrict__ xBCc, const float* __restrict__ Acum,
                                const float* __restrict__ prevS, float* __restrict__ y) {
  __shared__ __bf16 sCE[64 * SLD], sP[64 * SLD];
  __shared__ float sEA[64];
  int h = blockIdx.x, c = blockIdx.y, b = blockIdx.z, tid = threadIdx.x;
  const float* AcP = Acum + ((size_t)(b * NHEADS + h) * NCHUNK + c) * CHUNK;
  if (tid < 64) sEA[tid] = __expf(AcP[tid]);
  __syncthreads();
  const float* pp = prevS + ((size_t)(b * NCHUNK + c) * NHEADS + h) * (HEADDIM * D_STATE);
  for (int i = tid; i < 4096; i += 256) {
    int l = i >> 6, j = i & 63;
    size_t tok = (size_t)b * SEQLEN + c * CHUNK + l;
    sCE[l * SLD + j] = (__bf16)(xBCc[tok * CONV_DIM + D_INNER + D_STATE + j] * sEA[l]);
    sP[l * SLD + j]  = (__bf16)pp[l * 64 + j];     // here l indexes p, j indexes n
  }
  __syncthreads();
  int lane = tid & 31, w = tid >> 5, mt = w & 3, np = w >> 2;
  #pragma unroll
  for (int t = 0; t < 2; ++t) {
    int p = (np * 2 + t) * 16 + (lane & 15);
    f32x8 acc;
    #pragma unroll
    for (int vr = 0; vr < 8; ++vr) {
      int l = mt * 16 + vr + ((lane < 16) ? 0 : 8);
      size_t tok = (size_t)b * SEQLEN + c * CHUNK + l;
      acc[vr] = y[tok * D_INNER + h * HEADDIM + p];
    }
    for (int kk = 0; kk < 64; kk += 32) {
      bf16x16 af = frag_a(sCE, SLD, mt * 16, kk);
      acc = wmma_bf16(af, frag_b(sP, SLD, (np * 2 + t) * 16, kk), acc);
    }
    #pragma unroll
    for (int vr = 0; vr < 8; ++vr) {
      int l = mt * 16 + vr + ((lane < 16) ? 0 : 8);
      size_t tok = (size_t)b * SEQLEN + c * CHUNK + l;
      y[tok * D_INNER + h * HEADDIM + p] = acc[vr];
    }
  }
}

// ---------------- launcher ----------------------------------------------------

extern "C" void kernel_launch(void* const* d_in, const int* in_sizes, int n_in,
                              void* d_out, int out_size, void* d_ws, size_t ws_size,
                              hipStream_t stream) {
  (void)in_sizes; (void)n_in; (void)out_size; (void)ws_size;
  const float* x       = (const float*)d_in[0];
  const float* ln_w    = (const float*)d_in[1];
  const float* ln_b    = (const float*)d_in[2];
  const float* in_w    = (const float*)d_in[3];
  const float* conv_w  = (const float*)d_in[4];
  const float* conv_b  = (const float*)d_in[5];
  const float* dt_bias = (const float*)d_in[6];
  const float* A_log   = (const float*)d_in[7];
  const float* Dh      = (const float*)d_in[8];
  const float* rms_w   = (const float*)d_in[9];
  const float* out_w   = (const float*)d_in[10];
  float* out = (float*)d_out;

  char* wp = (char*)d_ws;
  auto alloc = [&](size_t bytes) -> void* {
    void* p = (void*)wp;
    wp += (bytes + 255) & ~(size_t)255;
    return p;
  };
  __bf16* hbf    = (__bf16*)alloc((size_t)NTOK * D_MODEL * 2);
  __bf16* inw_t  = (__bf16*)alloc((size_t)ZXP * D_MODEL * 2);       // [n][k], padded
  __bf16* outw_t = (__bf16*)alloc((size_t)D_MODEL * D_INNER * 2);   // [n][k]
  float*  zx     = (float*) alloc((size_t)NTOK * ZXP * 4);
  float*  xBCc   = (float*) alloc((size_t)NTOK * CONV_DIM * 4);
  float*  dtArr  = (float*) alloc((size_t)NTOK * NHEADS * 4);
  float*  Acum   = (float*) alloc((size_t)BATCH * NHEADS * NCHUNK * CHUNK * 4);
  float*  cdec   = (float*) alloc((size_t)BATCH * NHEADS * NCHUNK * 4);
  float*  states = (float*) alloc((size_t)BATCH * NCHUNK * NHEADS * HEADDIM * D_STATE * 4);
  float*  prevS  = (float*) alloc((size_t)BATCH * NCHUNK * NHEADS * HEADDIM * D_STATE * 4);
  float*  yb     = (float*) alloc((size_t)NTOK * D_INNER * 4);
  __bf16* gbf    = (__bf16*)alloc((size_t)NTOK * D_INNER * 2);

  // weight prep: transpose to N-major bf16, zero-pad N
  {
    long t1 = (long)ZXP * D_MODEL;
    castT_kernel<<<dim3((t1 + 255) / 256), dim3(256), 0, stream>>>(in_w, inw_t, D_MODEL, D_IN_PROJ, t1);
    long t2 = (long)D_MODEL * D_INNER;
    castT_kernel<<<dim3((t2 + 255) / 256), dim3(256), 0, stream>>>(out_w, outw_t, D_INNER, D_MODEL, t2);
  }
  // layernorm -> bf16
  ln_kernel<<<dim3(NTOK), dim3(256), 0, stream>>>(x, ln_w, ln_b, hbf);
  // in_proj GEMM: zx = hbf @ in_w  (padded N = ZXP)
  gemm_bf16_wmma<<<dim3(ZXP / GTN, NTOK / 64), dim3(256), 0, stream>>>(
      hbf, inw_t, nullptr, zx, D_MODEL, ZXP);
  // conv + silu
  conv_silu_kernel<<<dim3(((size_t)NTOK * CONV_DIM + 255) / 256), dim3(256), 0, stream>>>(
      zx, conv_w, conv_b, xBCc);
  // dt softplus + Acum cumsum + chunk decay
  dt_acum_kernel<<<dim3(NCHUNK, NHEADS, BATCH), dim3(CHUNK), 0, stream>>>(
      zx, dt_bias, A_log, dtArr, Acum, cdec);
  // per-chunk WMMA: Y_diag (+ D*x) and states
  ssd_chunk_kernel<<<dim3(NHEADS, NCHUNK, BATCH), dim3(256), 0, stream>>>(
      xBCc, dtArr, Acum, Dh, yb, states);
  // inter-chunk scan
  scan_kernel<<<dim3(NHEADS, BATCH), dim3(256), 0, stream>>>(states, cdec, prevS);
  // Y_off accumulate
  ssd_yoff_kernel<<<dim3(NHEADS, NCHUNK, BATCH), dim3(256), 0, stream>>>(
      xBCc, Acum, prevS, yb);
  // gating + RMSNorm -> bf16
  gate_kernel<<<dim3(NTOK), dim3(256), 0, stream>>>(yb, zx, rms_w, gbf);
  // out_proj GEMM with fused residual: out = x + gbf @ out_w
  gemm_bf16_wmma<<<dim3(D_MODEL / GTN, NTOK / 64), dim3(256), 0, stream>>>(
      gbf, outw_t, x, out, D_INNER, D_MODEL);
}
